// Decoder_35923106463917
// MI455X (gfx1250) — matrix operand order repaired and checked by reading, not compiled, not run on hardware
//
#include <hip/hip_runtime.h>
#include <stdint.h>

// ---------------------------------------------------------------------------
// Tacotron2-style decoder for MI455X (gfx1250, wave32, WMMA).
// Strategy: one persistent kernel for all 400 sequential steps.
//  * bf16-packed weights in WMMA B-fragment layout (L2-resident, ~36MB)
//  * activations packed into WMMA A-fragment layout each phase
//  * v_wmma_f32_16x16x32_bf16 for all LSTM-gate GEMMs (f32 accum)
//  * WGP-scope prefetch of the weight stream (locality 3 -> near caches)
//  * 3 device-wide barriers per step, WG roles overlap dependency chain
//  * teacher_forcing_ratio == 1 -> prenet(teacher frames) precomputed+packed
// ---------------------------------------------------------------------------

#define DEV __device__ __forceinline__

typedef __attribute__((ext_vector_type(16))) __bf16 v16bf;
typedef __attribute__((ext_vector_type(8)))  float  v8f;

constexpr int B_  = 32,  L_ = 200, E_ = 512, T_ = 400;
constexpr int D_  = 1024, P_ = 256, M_ = 80, A_ = 128, F_ = 32, KC_ = 31;
constexpr int ND  = 4 * D_;                 // 4096 gate columns
constexpr int KT_ATT_X = (P_ + E_) / 32;    // 24 k-tiles (frame||ctx)
constexpr int KT_D     = D_ / 32;           // 32 k-tiles
constexpr int KT_GEN_X = (D_ + E_) / 32;    // 48 k-tiles (h_att||ctx)
constexpr int KT_CTX   = E_ / 32;           // 16
constexpr int KT_FRM   = P_ / 32;           // 8
constexpr int NWG_GEMM = 64;                // one 16-col block per gate each
constexpr int NWG_ATT  = 32;                // one per batch row
constexpr int NWG      = NWG_GEMM + NWG_ATT;

// Prefetch the weight stream 2 fragments (2KB) ahead into near caches.
#define WPF(p) __builtin_prefetch((p) + 1024, 0, 3)

// ----------------------------- helpers -------------------------------------

DEV unsigned short f2bf(float f) {
  unsigned u = __builtin_bit_cast(unsigned, f);
  unsigned r = u + 0x7FFFu + ((u >> 16) & 1u);   // round-to-nearest-even
  return (unsigned short)(r >> 16);
}

// A-fragment (16x32 bf16) element offset for (row m in tile, k' = k%32),
// per CDNA5 ISA 7.12.2 16-bit A layout.
DEV int afrag_off(int m, int kq) {
  int lane = m + 16 * ((kq >> 3) & 1);
  int jj   = (kq >> 4) * 8 + (kq & 7);
  return lane * 16 + jj;
}

DEV v16bf ld_frag(const unsigned short* p) {
  int lane = threadIdx.x & 31;
  union { uint4 u[2]; v16bf v; } r;
  const uint4* q = (const uint4*)(p + (size_t)lane * 16);
  r.u[0] = q[0];
  r.u[1] = q[1];
  return r.v;
}

DEV v8f wmma_bf16(v16bf a, v16bf b, v8f c) {
  return __builtin_amdgcn_wmma_f32_16x16x32_bf16(false, a, false, b,
                                                 (short)0, c, false, false);
}

DEV float sigf(float x) { return 1.f / (1.f + __expf(-x)); }

// --------------------------- setup kernels ----------------------------------

// Pack W (row-major [N,K], used as B[k,n]=W[n,k]) into B-fragment layout:
// frag(nt,kt) holds 512 bf16; element (lane,jj): n=nt*16+lane%16,
// k=kt*32+(jj/8)*16+(lane/16)*8+jj%8.
__global__ void k_pack_b(const float* __restrict__ src,
                         unsigned short* __restrict__ dst,
                         int N, int Kdim, int KT) {
  int frag = blockIdx.x;
  int nt = frag / KT, kt = frag % KT;
  for (int e = threadIdx.x; e < 512; e += blockDim.x) {
    int lane = e >> 4, jj = e & 15;
    int n = nt * 16 + (lane & 15);
    int k = kt * 32 + (jj >> 3) * 16 + (lane >> 4) * 8 + (jj & 7);
    float v = 0.f;
    if (n < N && k < Kdim) v = src[(size_t)n * Kdim + k];
    dst[(size_t)frag * 512 + e] = f2bf(v);
  }
}

// prenet over teacher inputs for all (b,t); writes packed A-fragments.
__global__ void k_prenet(const float* __restrict__ target,
                         const float* __restrict__ w1, const float* __restrict__ b1,
                         const float* __restrict__ w2, const float* __restrict__ b2,
                         unsigned short* __restrict__ pa_frame) {
  int bt = blockIdx.x;
  int b = bt / T_, t = bt % T_;
  __shared__ float xs[M_];
  __shared__ float x1[P_];
  int tid = threadIdx.x;
  if (tid < M_)
    xs[tid] = (t == 0) ? 0.f : target[((size_t)b * M_ + tid) * T_ + (t - 1)];
  __syncthreads();
  {
    float s = b1[tid];
    for (int m = 0; m < M_; ++m) s += xs[m] * w1[m * P_ + tid];
    x1[tid] = fmaxf(s, 0.f);
  }
  __syncthreads();
  float s = b2[tid];
  for (int q = 0; q < P_; ++q) s += x1[q] * w2[q * P_ + tid];
  s = fmaxf(s, 0.f);
  int k = tid, kt = k >> 5, kq = k & 31, mt = b >> 4;
  pa_frame[(size_t)(((t * 2 + mt) * KT_FRM) + kt) * 512 + afrag_off(b & 15, kq)] =
      f2bf(s);
}

// processed memory pm[b,l,a] = enc[b,l,:] @ wm
__global__ void k_pm(const float* __restrict__ enc, const float* __restrict__ wm,
                     float* __restrict__ pm) {
  int bl = blockIdx.x, a = threadIdx.x;
  float s = 0.f;
  const float* ep = enc + (size_t)bl * E_;
  for (int e = 0; e < E_; ++e) s += ep[e] * wm[e * A_ + a];
  pm[(size_t)bl * A_ + a] = s;
}

__global__ void k_zero(unsigned* __restrict__ p, size_t nwords) {
  size_t i = (size_t)blockIdx.x * blockDim.x + threadIdx.x;
  size_t stride = (size_t)gridDim.x * blockDim.x;
  for (; i < nwords; i += stride) p[i] = 0u;
}

// --------------------------- decoder kernel ---------------------------------

struct DecParams {
  const float* enc;
  const int* lens;
  const float *att_bih, *att_bhh, *gen_bih, *gen_bhh;
  const float *wq, *loc_conv, *wloc, *v_att, *frame_w, *frame_b, *stop_w, *stop_b;
  const unsigned short *att_wih_p, *att_whh_p, *gen_wih_p, *gen_whh_p, *pa_frame;
  const float* pm;
  float* locA;
  float *h_att[2], *c_att[2], *h_gen[2], *c_gen[2], *ctx[2], *wbuf[2], *cum[2];
  unsigned short *pa_hatt[2], *pa_hgen[2], *pa_ctx[2];
  unsigned *bar_cnt, *bar_gen;
  float *spec_out, *stop_out, *align_out;
};

DEV void gbar(unsigned* cnt, unsigned* gen, unsigned& lg) {
  __threadfence();
  __syncthreads();
  lg++;
  if (threadIdx.x == 0) {
    unsigned prev = atomicAdd(cnt, 1u);
    if (prev == (unsigned)(NWG - 1)) {
      atomicExch(cnt, 0u);
      __threadfence();
      atomicAdd(gen, 1u);
    } else {
      while (atomicAdd(gen, 0u) < lg) __builtin_amdgcn_s_sleep(2);
    }
  }
  __syncthreads();
  __threadfence();
}

// LSTM pointwise update from gates staged in LDS [4][32][16]; writes f32 h,
// packed-A h fragments, and new c.  wg owns columns [wg*16, wg*16+16).
DEV void lstm_ew(const float* lds, int wg, int tid,
                 const float* bih, const float* bhh,
                 const float* c_r, float* c_w, float* h_w,
                 unsigned short* pa_w) {
#pragma unroll
  for (int rep = 0; rep < 2; ++rep) {
    int idx = tid + rep * 256;
    int m = idx >> 4, dl = idx & 15;
    int d = wg * 16 + dl;
    float gi = lds[(0 * 32 + m) * 16 + dl] + bih[0 * D_ + d] + bhh[0 * D_ + d];
    float gf = lds[(1 * 32 + m) * 16 + dl] + bih[1 * D_ + d] + bhh[1 * D_ + d];
    float gg = lds[(2 * 32 + m) * 16 + dl] + bih[2 * D_ + d] + bhh[2 * D_ + d];
    float go = lds[(3 * 32 + m) * 16 + dl] + bih[3 * D_ + d] + bhh[3 * D_ + d];
    float c2 = sigf(gf) * c_r[m * D_ + d] + sigf(gi) * tanhf(gg);
    float h  = sigf(go) * tanhf(c2);
    c_w[m * D_ + d] = c2;
    h_w[m * D_ + d] = h;
    pa_w[(size_t)(((m >> 4) * KT_D) + (d >> 5)) * 512 + afrag_off(m & 15, d & 31)] =
        f2bf(h);
  }
}

__global__ __launch_bounds__(256, 1) void k_decoder(DecParams pr) {
  __shared__ float lds[6656];
  const int wg   = blockIdx.x;
  const int tid  = threadIdx.x;
  const int lane = tid & 31;
  const int wv   = tid >> 5;         // 8 waves / WG
  const bool is_gemm = (wg < NWG_GEMM);
  const int bb = wg - NWG_GEMM;      // batch row for attention role
  unsigned barlg = 0;

  for (int t = 0; t < T_; ++t) {
    const int rp = t & 1, wc = rp ^ 1;

    // ---------------- Phase A: att gates GEMM || location features ---------
    if (is_gemm) {
      const int g = wv >> 1, mt = wv & 1;   // wave -> (gate, m-tile)
      v8f acc = {0.f, 0.f, 0.f, 0.f, 0.f, 0.f, 0.f, 0.f};
      for (int kt = 0; kt < KT_ATT_X; ++kt) {               // x = [frame||ctx]
        const unsigned short* ap =
            (kt < KT_FRM)
                ? pr.pa_frame + (size_t)(((t * 2 + mt) * KT_FRM) + kt) * 512
                : pr.pa_ctx[rp] + (size_t)((mt * KT_CTX) + (kt - KT_FRM)) * 512;
        const unsigned short* bp =
            pr.att_wih_p + (size_t)(((g * 64 + wg) * KT_ATT_X) + kt) * 512;
        WPF(bp);
        acc = wmma_bf16(ld_frag(ap), ld_frag(bp), acc);
      }
      for (int kt = 0; kt < KT_D; ++kt) {                   // + h_att(t-1)@Whh
        const unsigned short* ap =
            pr.pa_hatt[rp] + (size_t)((mt * KT_D) + kt) * 512;
        const unsigned short* bp =
            pr.att_whh_p + (size_t)(((g * 64 + wg) * KT_D) + kt) * 512;
        WPF(bp);
        acc = wmma_bf16(ld_frag(ap), ld_frag(bp), acc);
      }
      {
        int mrow = mt * 16 + 8 * (lane >> 4);
        int col  = lane & 15;
#pragma unroll
        for (int r = 0; r < 8; ++r)
          lds[(g * 32 + mrow + r) * 16 + col] = acc[r];
      }
      __syncthreads();
      lstm_ew(lds, wg, tid, pr.att_bih, pr.att_bhh,
              pr.c_att[rp], pr.c_att[wc], pr.h_att[wc], pr.pa_hatt[wc]);
    } else {
      // location conv + wloc projection (depends only on t-1 attention state)
      const float* pw = pr.wbuf[rp] + bb * L_;
      const float* cw = pr.cum[rp] + bb * L_;
      float* loc = lds;  // [F_][L_]
      for (int idx = tid; idx < F_ * L_; idx += 256) {
        int f = idx / L_, l = idx % L_;
        const float* wk = pr.loc_conv + (f * 2) * KC_;
        float s = 0.f;
        for (int kk = 0; kk < KC_; ++kk) {
          int li = l + kk - KC_ / 2;
          if (li >= 0 && li < L_) s += pw[li] * wk[kk] + cw[li] * wk[KC_ + kk];
        }
        loc[f * L_ + l] = s;
      }
      __syncthreads();
      for (int idx = tid; idx < L_ * A_; idx += 256) {
        int l = idx / A_, a = idx % A_;
        float s = 0.f;
        for (int f = 0; f < F_; ++f) s += loc[f * L_ + l] * pr.wloc[f * A_ + a];
        pr.locA[((size_t)bb * L_ + l) * A_ + a] = s;
      }
      __syncthreads();
    }
    gbar(pr.bar_cnt, pr.bar_gen, barlg);   // h_att(t) now visible

    // ------- Phase B: attention (per-batch) || gen gates h-dependent part ---
    v8f accg = {0.f, 0.f, 0.f, 0.f, 0.f, 0.f, 0.f, 0.f};
    if (is_gemm) {
      const int g = wv >> 1, mt = wv & 1;
      for (int kt = 0; kt < KT_D; ++kt) {                   // h_att(t) @ Wih
        const unsigned short* ap =
            pr.pa_hatt[wc] + (size_t)((mt * KT_D) + kt) * 512;
        const unsigned short* bp =
            pr.gen_wih_p + (size_t)(((g * 64 + wg) * KT_GEN_X) + kt) * 512;
        WPF(bp);
        accg = wmma_bf16(ld_frag(ap), ld_frag(bp), accg);
      }
      for (int kt = 0; kt < KT_D; ++kt) {                   // h_gen(t-1) @ Whh
        const unsigned short* ap =
            pr.pa_hgen[rp] + (size_t)((mt * KT_D) + kt) * 512;
        const unsigned short* bp =
            pr.gen_whh_p + (size_t)(((g * 64 + wg) * KT_D) + kt) * 512;
        WPF(bp);
        accg = wmma_bf16(ld_frag(ap), ld_frag(bp), accg);
      }
    } else {
      float* q  = lds;        // [128]
      float* en = lds + 128;  // [200]
      float* rd = lds + 336;  // [256]
      const float* h = pr.h_att[wc] + bb * D_;
      for (int a = tid; a < A_; a += 256) {
        float s = 0.f;
        for (int k = 0; k < D_; ++k) s += h[k] * pr.wq[k * A_ + a];
        q[a] = s;
      }
      __syncthreads();
      int len = pr.lens[bb];
      for (int l = tid; l < L_; l += 256) {
        const float* la  = pr.locA + ((size_t)bb * L_ + l) * A_;
        const float* pmp = pr.pm + ((size_t)bb * L_ + l) * A_;
        float s = 0.f;
        for (int a = 0; a < A_; ++a)
          s += tanhf(q[a] + la[a] + pmp[a]) * pr.v_att[a];
        en[l] = (l < len) ? s : -1e9f;
      }
      __syncthreads();
      float lm = -3.4e38f;
      for (int l = tid; l < L_; l += 256) lm = fmaxf(lm, en[l]);
      rd[tid] = lm;
      __syncthreads();
      for (int s = 128; s > 0; s >>= 1) {
        if (tid < s) rd[tid] = fmaxf(rd[tid], rd[tid + s]);
        __syncthreads();
      }
      float mx = rd[0];
      __syncthreads();
      float ls = 0.f;
      for (int l = tid; l < L_; l += 256) ls += __expf(en[l] - mx);
      rd[tid] = ls;
      __syncthreads();
      for (int s = 128; s > 0; s >>= 1) {
        if (tid < s) rd[tid] += rd[tid + s];
        __syncthreads();
      }
      float inv = 1.f / rd[0];
      __syncthreads();
      for (int l = tid; l < L_; l += 256) {
        float w = __expf(en[l] - mx) * inv;
        en[l] = w;
        pr.wbuf[wc][bb * L_ + l] = w;
        pr.cum[wc][bb * L_ + l] = pr.cum[rp][bb * L_ + l] + w;
        pr.align_out[((size_t)bb * T_ + t) * L_ + l] = w;
      }
      __syncthreads();
      for (int e = tid; e < E_; e += 256) {                 // context
        const float* ep = pr.enc + (size_t)bb * L_ * E_ + e;
        float s = 0.f;
        for (int l = 0; l < L_; ++l) s += en[l] * ep[(size_t)l * E_];
        pr.ctx[wc][bb * E_ + e] = s;
        pr.pa_ctx[wc][(size_t)(((bb >> 4) * KT_CTX) + (e >> 5)) * 512 +
                      afrag_off(bb & 15, e & 31)] = f2bf(s);
      }
      __syncthreads();
    }
    gbar(pr.bar_cnt, pr.bar_gen, barlg);   // context(t) now visible

    // ---------------- Phase C: gen gates context part + LSTM update ---------
    if (is_gemm) {
      const int g = wv >> 1, mt = wv & 1;
      for (int kt = 0; kt < KT_CTX; ++kt) {
        const unsigned short* ap =
            pr.pa_ctx[wc] + (size_t)((mt * KT_CTX) + kt) * 512;
        const unsigned short* bp =
            pr.gen_wih_p + (size_t)(((g * 64 + wg) * KT_GEN_X) + KT_D + kt) * 512;
        WPF(bp);
        accg = wmma_bf16(ld_frag(ap), ld_frag(bp), accg);
      }
      {
        int mrow = mt * 16 + 8 * (lane >> 4);
        int col  = lane & 15;
#pragma unroll
        for (int r = 0; r < 8; ++r)
          lds[(g * 32 + mrow + r) * 16 + col] = accg[r];
      }
      __syncthreads();
      lstm_ew(lds, wg, tid, pr.gen_bih, pr.gen_bhh,
              pr.c_gen[rp], pr.c_gen[wc], pr.h_gen[wc], pr.pa_hgen[wc]);
    }
    gbar(pr.bar_cnt, pr.bar_gen, barlg);   // h_gen(t) now visible

    // ---------------- Phase D: frame / stop projections ---------------------
    if (!is_gemm) {
      const float* hg = pr.h_gen[wc] + bb * D_;
      const float* cx = pr.ctx[wc] + bb * E_;
      if (tid < M_) {
        float s = pr.frame_b[tid];
        for (int k = 0; k < D_; ++k) s += hg[k] * pr.frame_w[k * M_ + tid];
        for (int k = 0; k < E_; ++k) s += cx[k] * pr.frame_w[(D_ + k) * M_ + tid];
        pr.spec_out[((size_t)bb * T_ + t) * M_ + tid] = s;
      } else if (tid == M_) {
        float s = pr.stop_b[0];
        for (int k = 0; k < D_; ++k) s += hg[k] * pr.stop_w[k];
        for (int k = 0; k < E_; ++k) s += cx[k] * pr.stop_w[D_ + k];
        pr.stop_out[(size_t)bb * T_ + t] = s;
      }
    }
    // no barrier needed: next phase A touches only t-disjoint buffers
  }
}

// ------------------------------ launch --------------------------------------

extern "C" void kernel_launch(void* const* d_in, const int* in_sizes, int n_in,
                              void* d_out, int out_size, void* d_ws,
                              size_t ws_size, hipStream_t stream) {
  const float* enc      = (const float*)d_in[0];
  const int*   lens     = (const int*)d_in[1];
  const float* target   = (const float*)d_in[2];
  // d_in[3] teacher_forcing_ratio == 1 -> teacher forcing always on.
  const float* pre_w1   = (const float*)d_in[4];
  const float* pre_b1   = (const float*)d_in[5];
  const float* pre_w2   = (const float*)d_in[6];
  const float* pre_b2   = (const float*)d_in[7];
  const float* att_wih  = (const float*)d_in[8];
  const float* att_whh  = (const float*)d_in[9];
  const float* att_bih  = (const float*)d_in[10];
  const float* att_bhh  = (const float*)d_in[11];
  const float* gen_wih  = (const float*)d_in[12];
  const float* gen_whh  = (const float*)d_in[13];
  const float* gen_bih  = (const float*)d_in[14];
  const float* gen_bhh  = (const float*)d_in[15];
  const float* wq       = (const float*)d_in[16];
  const float* wm       = (const float*)d_in[17];
  const float* loc_conv = (const float*)d_in[18];
  const float* wloc     = (const float*)d_in[19];
  const float* v_att    = (const float*)d_in[20];
  const float* frame_w  = (const float*)d_in[21];
  const float* frame_b  = (const float*)d_in[22];
  const float* stop_w   = (const float*)d_in[23];
  const float* stop_b   = (const float*)d_in[24];

  uint8_t* base = (uint8_t*)d_ws;
  size_t off = 0;
  auto take = [&](size_t bytes) -> uint8_t* {
    uint8_t* r = base + off;
    off = (off + bytes + 255) & ~(size_t)255;
    return r;
  };

  unsigned short* att_wih_p = (unsigned short*)take((size_t)256 * KT_ATT_X * 512 * 2);
  unsigned short* att_whh_p = (unsigned short*)take((size_t)256 * KT_D * 512 * 2);
  unsigned short* gen_wih_p = (unsigned short*)take((size_t)256 * KT_GEN_X * 512 * 2);
  unsigned short* gen_whh_p = (unsigned short*)take((size_t)256 * KT_D * 512 * 2);
  unsigned short* pa_frame  = (unsigned short*)take((size_t)T_ * 2 * KT_FRM * 512 * 2);
  float* pm   = (float*)take((size_t)B_ * L_ * A_ * 4);
  float* locA = (float*)take((size_t)B_ * L_ * A_ * 4);

  uint8_t* state_begin = base + off;
  DecParams pr;
  pr.enc = enc; pr.lens = lens;
  pr.att_bih = att_bih; pr.att_bhh = att_bhh;
  pr.gen_bih = gen_bih; pr.gen_bhh = gen_bhh;
  pr.wq = wq; pr.loc_conv = loc_conv; pr.wloc = wloc; pr.v_att = v_att;
  pr.frame_w = frame_w; pr.frame_b = frame_b; pr.stop_w = stop_w; pr.stop_b = stop_b;
  pr.att_wih_p = att_wih_p; pr.att_whh_p = att_whh_p;
  pr.gen_wih_p = gen_wih_p; pr.gen_whh_p = gen_whh_p;
  pr.pa_frame = pa_frame; pr.pm = pm; pr.locA = locA;
  for (int s = 0; s < 2; ++s) {
    pr.h_att[s] = (float*)take((size_t)B_ * D_ * 4);
    pr.c_att[s] = (float*)take((size_t)B_ * D_ * 4);
    pr.h_gen[s] = (float*)take((size_t)B_ * D_ * 4);
    pr.c_gen[s] = (float*)take((size_t)B_ * D_ * 4);
    pr.ctx[s]   = (float*)take((size_t)B_ * E_ * 4);
    pr.wbuf[s]  = (float*)take((size_t)B_ * L_ * 4);
    pr.cum[s]   = (float*)take((size_t)B_ * L_ * 4);
    pr.pa_hatt[s] = (unsigned short*)take((size_t)2 * KT_D * 512 * 2);
    pr.pa_hgen[s] = (unsigned short*)take((size_t)2 * KT_D * 512 * 2);
    pr.pa_ctx[s]  = (unsigned short*)take((size_t)2 * KT_CTX * 512 * 2);
  }
  unsigned* bar = (unsigned*)take(256);
  pr.bar_cnt = bar; pr.bar_gen = bar + 1;
  size_t state_bytes = (size_t)((base + off) - state_begin);

  pr.spec_out  = (float*)d_out;
  pr.stop_out  = pr.spec_out + (size_t)B_ * T_ * M_;
  pr.align_out = pr.stop_out + (size_t)B_ * T_;

  // setup: pack weights (bf16 B-fragments), prenet+pack, processed memory,
  // zero ping-pong state + barrier words.
  k_pack_b<<<256 * KT_ATT_X, 256, 0, stream>>>(att_wih, att_wih_p, ND, P_ + E_, KT_ATT_X);
  k_pack_b<<<256 * KT_D,     256, 0, stream>>>(att_whh, att_whh_p, ND, D_, KT_D);
  k_pack_b<<<256 * KT_GEN_X, 256, 0, stream>>>(gen_wih, gen_wih_p, ND, D_ + E_, KT_GEN_X);
  k_pack_b<<<256 * KT_D,     256, 0, stream>>>(gen_whh, gen_whh_p, ND, D_, KT_D);
  k_prenet<<<B_ * T_, P_, 0, stream>>>(target, pre_w1, pre_b1, pre_w2, pre_b2, pa_frame);
  k_pm<<<B_ * L_, A_, 0, stream>>>(enc, wm, pm);
  k_zero<<<512, 256, 0, stream>>>((unsigned*)state_begin, state_bytes / 4);

  // persistent sequential decoder: 96 co-resident workgroups, 400 steps.
  k_decoder<<<NWG, 256, 0, stream>>>(pr);
}
